// Generator_PS_70403103916218
// MI455X (gfx1250) — matrix operand order repaired and checked by reading, not compiled, not run on hardware
//
#include <hip/hip_runtime.h>
#include <hip/hip_bf16.h>
#include <cstdio>
#include <cstring>

// ---------------------------------------------------------------------------
// CDNA5 WMMA types (wave32): V_WMMA_F32_16X16X32_BF16
//   A: 16x32 bf16 -> 16 bf16/lane, B: 32x16 bf16 -> 16 bf16/lane, C/D: 8 f32/lane
// ---------------------------------------------------------------------------
typedef __attribute__((ext_vector_type(16))) __bf16 v16bf;
typedef __attribute__((ext_vector_type(8)))  __bf16 v8bf;
typedef __attribute__((ext_vector_type(8)))  float  v8f;

__device__ __forceinline__ __bf16 f2bf(float f) {
  unsigned u = __builtin_bit_cast(unsigned, f);
  u += 0x7FFFu + ((u >> 16) & 1u);              // round-to-nearest-even
  unsigned short h = (unsigned short)(u >> 16);
  return __builtin_bit_cast(__bf16, h);
}

// A-fragment K offset for element i (ISA 16-bit A 16x32 layout).
__device__ __forceinline__ int a_koff(int i, int half) {
  const int v = i >> 1;
  return ((v < 4) ? (2 * v) : (16 + 2 * (v - 4))) + (i & 1) + half * 8;
}

// Async DMA: 16 bytes global -> LDS (GLOBAL_LOAD_ASYNC_TO_LDS_B128, ASYNCcnt).
// LDS byte offset = low 32 bits of the generic pointer.
__device__ __forceinline__ void async_ld16(void* lds, const void* gmem) {
  const unsigned loff = (unsigned)(unsigned long long)lds;
  asm volatile("global_load_async_to_lds_b128 %0, %1, off"
               :: "v"(loff), "v"(gmem) : "memory");
}
__device__ __forceinline__ void wait_asynccnt0() {
  asm volatile("s_wait_asynccnt 0x0" ::: "memory");
}

// ---------------------------------------------------------------------------
// Weight packer: f32 conv weights -> bf16 B-fragments.
// Bp[(((kc*NT + nt)*32 + lane)*16) + i], K = kc*32 + (i<8 ? i : i+8) + 8*(lane>=16),
// oc = nt*16 + (lane&15). mode 1 = ConvTranspose2d flip/transpose.
// ---------------------------------------------------------------------------
__global__ void pack_weights_kernel(const float* __restrict__ w, __bf16* __restrict__ bp,
                                    int Cin, int Cout, int ksz, int mode, int NT,
                                    int total) {
  const int idx = blockIdx.x * blockDim.x + threadIdx.x;
  if (idx >= total) return;
  const int i = idx & 15, lane = (idx >> 4) & 31, t = idx >> 9;
  const int nt = t % NT, kc = t / NT;
  const int half = lane >> 4, l16 = lane & 15;
  const int K = kc * 32 + (i < 8 ? i : i + 8) + 8 * half;
  const int oc = nt * 16 + l16;
  const int kk2 = ksz * ksz, Ktot = Cin * kk2;
  float v = 0.f;
  if (K < Ktot && oc < Cout) {
    if (mode == 0) {
      v = w[(size_t)oc * Ktot + K];
    } else {
      const int ic = K / kk2, rem = K - ic * kk2;
      const int kh = rem / ksz, kw = rem - kh * ksz;
      v = w[(((size_t)ic * Cout + oc) * ksz + (ksz - 1 - kh)) * ksz + (ksz - 1 - kw)];
    }
  }
  bp[idx] = f2bf(v);
}

// ---------------------------------------------------------------------------
// Implicit-GEMM conv / lhs-dilated deconv. Block = 128 thr (4 waves),
// tile 64M x 64N, A (64x32 im2col bf16) built in LDS fragment-order per
// K-step; each wave does 4 WMMAs per 32B B-fragment load. M % 64 == 0.
// ---------------------------------------------------------------------------
template <int KSZ, int MODE>
__global__ __launch_bounds__(128)
void conv_wmma_kernel(const float* __restrict__ in, const __bf16* __restrict__ Bp,
                      float* __restrict__ out, int Cin, int Cout,
                      int Hin, int Win, int Hout, int Wout, int stride, int pad) {
  constexpr int kk2 = KSZ * KSZ;
  const int Ktot = Cin * kk2;
  const int Kch  = (Ktot + 31) >> 5;
  const int M    = Hout * Wout;
  const int tid  = threadIdx.x;

  const int bLane = tid & 31, bMt = tid >> 5;
  const int bHalf = bLane >> 4, bL16 = bLane & 15;
  const int rowB  = blockIdx.x * 64 + bMt * 16 + bL16;
  const int oh    = rowB / Wout, ow = rowB % Wout;

  const int wave = tid >> 5, lane = tid & 31;
  const int half = lane >> 4, l16 = lane & 15;
  const int NT   = gridDim.y * 4;
  const int ntG  = blockIdx.y * 4 + wave;
  const int col  = ntG * 16 + l16;

  __shared__ __bf16 shA[64 * 32];  // 4 KB, fragment order
  v8f acc[4] = {};

  for (int kc = 0; kc < Kch; ++kc) {
    const int k0 = kc * 32;
    v16bf aBuild;
#pragma unroll
    for (int i = 0; i < 16; ++i) {
      const int K = k0 + a_koff(i, bHalf);
      float v = 0.f;
      if (K < Ktot) {
        const int ic = K / kk2, rem = K - ic * kk2;
        const int kh = rem / KSZ, kw = rem - kh * KSZ;
        const int ihD = oh * stride - pad + kh;
        const int iwD = ow * stride - pad + kw;
        if (MODE == 0) {
          if (ihD >= 0 && ihD < Hin && iwD >= 0 && iwD < Win)
            v = in[((size_t)ic * Hin + ihD) * Win + iwD];
        } else {  // lhs_dilation = 2
          if (ihD >= 0 && iwD >= 0 && !(ihD & 1) && !(iwD & 1)) {
            const int ih = ihD >> 1, iw = iwD >> 1;
            if (ih < Hin && iw < Win) v = in[((size_t)ic * Hin + ih) * Win + iw];
          }
        }
      }
      aBuild[i] = f2bf(v);
    }
    *reinterpret_cast<v16bf*>(&shA[(bMt * 32 + bLane) * 16]) = aBuild;
    __syncthreads();

    const v16bf bV =
        *reinterpret_cast<const v16bf*>(Bp + ((size_t)(kc * NT + ntG) * 32 + lane) * 16);
    if (kc + 1 < Kch)
      __builtin_prefetch(Bp + ((size_t)((kc + 1) * NT + ntG) * 32 + lane) * 16, 0, 0);

#pragma unroll
    for (int mt = 0; mt < 4; ++mt) {
      const v16bf aV = *reinterpret_cast<const v16bf*>(&shA[(mt * 32 + lane) * 16]);
      acc[mt] = __builtin_amdgcn_wmma_f32_16x16x32_bf16(false, aV, false, bV,
                                                        (short)0, acc[mt], false, false);
    }
    __syncthreads();
  }

  if (col < Cout) {
#pragma unroll
    for (int mt = 0; mt < 4; ++mt)
#pragma unroll
      for (int r = 0; r < 8; ++r) {
        const int mr = blockIdx.x * 64 + mt * 16 + r + 8 * half;
        if (mr < M) out[(size_t)col * M + mr] = acc[mt][r];
      }
  }
}

// ---------------------------------------------------------------------------
// Attention GEMM: C[M,N] = A_bf16[M,lda] * Bfrag_bf16. A tiles staged with
// async global->LDS DMA (ASYNCcnt), double-buffered: one barrier per K-step,
// DMA of chunk kc+1 overlaps the 16 WMMAs of chunk kc.
// ---------------------------------------------------------------------------
__global__ __launch_bounds__(128)
void gemm_bf16_wmma_kernel(const __bf16* __restrict__ A, const __bf16* __restrict__ Bp,
                           float* __restrict__ C, int M, int N, int Kch, int lda) {
  const int tid = threadIdx.x;
  const int bLane = tid & 31, bMt = tid >> 5;
  const int bHalf = bLane >> 4, bL16 = bLane & 15;
  const int rowB  = blockIdx.x * 64 + bMt * 16 + bL16;

  const int wave = tid >> 5, lane = tid & 31;
  const int half = lane >> 4, l16 = lane & 15;
  const int NT   = gridDim.y * 4;
  const int ntG  = blockIdx.y * 4 + wave;
  const int col  = ntG * 16 + l16;

  __shared__ __bf16 shA[2][64 * 32];  // 2 x 4 KB, fragment order
  const int frag = (bMt * 32 + bLane) * 16;
  const __bf16* arow = A + (size_t)rowB * lda + bHalf * 8;

  // stage chunk 0 into buffer 0
  async_ld16(&shA[0][frag],     arow);
  async_ld16(&shA[0][frag + 8], arow + 16);

  v8f acc[4] = {};
  for (int kc = 0; kc < Kch; ++kc) {
    wait_asynccnt0();       // chunk kc landed in LDS
    __syncthreads();        // visible to all; previous buffer free to overwrite
    const int cur = kc & 1;
    if (kc + 1 < Kch) {     // kick off DMA for next chunk into other buffer
      const __bf16* an = arow + (size_t)(kc + 1) * 32;
      async_ld16(&shA[cur ^ 1][frag],     an);
      async_ld16(&shA[cur ^ 1][frag + 8], an + 16);
    }
    const v16bf bV =
        *reinterpret_cast<const v16bf*>(Bp + ((size_t)(kc * NT + ntG) * 32 + lane) * 16);
    if (kc + 1 < Kch)
      __builtin_prefetch(Bp + ((size_t)((kc + 1) * NT + ntG) * 32 + lane) * 16, 0, 0);

#pragma unroll
    for (int mt = 0; mt < 4; ++mt) {
      const v16bf aV = *reinterpret_cast<const v16bf*>(&shA[cur][(mt * 32 + lane) * 16]);
      acc[mt] = __builtin_amdgcn_wmma_f32_16x16x32_bf16(false, aV, false, bV,
                                                        (short)0, acc[mt], false, false);
    }
  }

  if (col < N) {
#pragma unroll
    for (int mt = 0; mt < 4; ++mt)
#pragma unroll
      for (int r = 0; r < 8; ++r) {
        const int mr = blockIdx.x * 64 + mt * 16 + r + 8 * half;
        if (mr < M) C[(size_t)mr * N + col] = acc[mt][r];
      }
  }
}

// ---------------------------------------------------------------------------
// InstanceNorm (+optional ReLU, +optional residual add): one block / channel.
// ---------------------------------------------------------------------------
__global__ __launch_bounds__(256)
void inorm_kernel(const float* __restrict__ x, const float* __restrict__ g,
                  const float* __restrict__ b, const float* __restrict__ res,
                  float* __restrict__ out, int HW, int doRelu) {
  __shared__ float sSum[256], sSq[256];
  const int c = blockIdx.x, tid = threadIdx.x;
  const float* xc = x + (size_t)c * HW;
  float s = 0.f, q = 0.f;
  for (int i = tid; i < HW; i += 256) { float v = xc[i]; s += v; q += v * v; }
  sSum[tid] = s; sSq[tid] = q; __syncthreads();
  for (int off = 128; off > 0; off >>= 1) {
    if (tid < off) { sSum[tid] += sSum[tid + off]; sSq[tid] += sSq[tid + off]; }
    __syncthreads();
  }
  const float mean = sSum[0] / (float)HW;
  const float var  = sSq[0] / (float)HW - mean * mean;
  const float inv  = rsqrtf(var + 1e-5f);
  const float gg = g[c], bb = b[c];
  float* oc_ = out + (size_t)c * HW;
  const float* rc = res ? res + (size_t)c * HW : nullptr;
  for (int i = tid; i < HW; i += 256) {
    float v = (xc[i] - mean) * inv * gg + bb;
    if (doRelu) v = fmaxf(v, 0.f);
    if (rc) v += rc[i];
    oc_[i] = v;
  }
}

// beta/gamma 1x1 conv (256 -> 1) over feat_ref, both at once.
__global__ void betagamma_kernel(const float* __restrict__ feat,
                                 const float* __restrict__ aw1, const float* __restrict__ ab1,
                                 const float* __restrict__ aw2, const float* __restrict__ ab2,
                                 float* __restrict__ beta, float* __restrict__ gama,
                                 int C, int HW) {
  const int p = blockIdx.x * blockDim.x + threadIdx.x;
  if (p >= HW) return;
  float s1 = 0.f, s2 = 0.f;
  for (int c = 0; c < C; ++c) {
    const float v = feat[(size_t)c * HW + p];
    s1 += v * aw1[c]; s2 += v * aw2[c];
  }
  beta[p] = s1 + ab1[0];
  gama[p] = s2 + ab2[0];
}

// src matrix (raw-reshape of concat) as row-major bf16 [HW x Kpad], zero-padded.
__global__ void build_catA_kernel(const float* __restrict__ feat, const float* __restrict__ lm,
                                  __bf16* __restrict__ A, int HW, int Kpad) {
  const int idx = blockIdx.x * blockDim.x + threadIdx.x;
  if (idx >= HW * Kpad) return;
  const int row = idx / Kpad, k = idx - row * Kpad;
  float v = 0.f;
  if (k < 392) {
    const int e = row * 392 + k;          // raw reshape semantics
    const int ch = e >> 12, p = e & 4095; // 4096 = HW
    v = (ch < 256) ? 0.01f * feat[(size_t)ch * HW + p] : lm[(size_t)(ch - 256) * HW + p];
  }
  A[idx] = f2bf(v);
}

// ref matrix (concat channel-major) packed straight into B-fragment layout.
__global__ void build_catB_frag_kernel(const float* __restrict__ feat,
                                       const float* __restrict__ lm,
                                       __bf16* __restrict__ bp, int HW, int NT, int total) {
  const int idx = blockIdx.x * blockDim.x + threadIdx.x;
  if (idx >= total) return;
  const int i = idx & 15, lane = (idx >> 4) & 31, t = idx >> 9;
  const int nt = t % NT, kc = t / NT;
  const int half = lane >> 4, l16 = lane & 15;
  const int ch = kc * 32 + (i < 8 ? i : i + 8) + 8 * half;   // K = channel of ref matrix
  const int j  = nt * 16 + l16;             // N = pixel
  float v = 0.f;
  if (ch < 256)      v = 0.01f * feat[(size_t)ch * HW + j];
  else if (ch < 392) v = lm[(size_t)(ch - 256) * HW + j];
  bp[idx] = f2bf(v);
}

// One-pass online-softmax column stats: cmax[j] = max_i A[i,j],
// cscale[j] = mask_j / sum_i exp(A[i,j]-cmax[j]).
__global__ __launch_bounds__(256)
void colstats_kernel(const float* __restrict__ A, const int* __restrict__ msrc,
                     const int* __restrict__ mref, float* __restrict__ cmax,
                     float* __restrict__ cscale, int HW) {
  __shared__ float sm[256], ss[256];
  const int j = blockIdx.x, tid = threadIdx.x;
  float m = -3.4e38f, s = 0.f;
  for (int i = tid; i < HW; i += 256) {
    const float a = A[(size_t)i * HW + j];
    if (a > m) { s *= __expf(m - a); m = a; }
    s += __expf(a - m);
  }
  sm[tid] = m; ss[tid] = s; __syncthreads();
  for (int off = 128; off > 0; off >>= 1) {
    if (tid < off) {
      const float m1 = sm[tid], s1 = ss[tid];
      const float m2 = sm[tid + off], s2 = ss[tid + off];
      const float mm = fmaxf(m1, m2);
      ss[tid] = s1 * __expf(m1 - mm) + s2 * __expf(m2 - mm);
      sm[tid] = mm;
    }
    __syncthreads();
  }
  if (tid == 0) {
    cmax[j] = sm[0];
    cscale[j] = ((mref[j] == msrc[j]) ? 1.f : 0.f) / ss[0];
  }
}

// Fused: beta_hat[i] = sum_j exp(A[i,j]-cmax[j])*cscale[j]*beta[j] (and gamma).
__global__ __launch_bounds__(256)
void attvec_fused_kernel(const float* __restrict__ A, const float* __restrict__ cmax,
                         const float* __restrict__ cscale, const float* __restrict__ beta,
                         const float* __restrict__ gama, float* __restrict__ bh,
                         float* __restrict__ gh, int HW) {
  __shared__ float r1[256], r2[256];
  const int i = blockIdx.x, tid = threadIdx.x;
  const float* Ar = A + (size_t)i * HW;
  float s1 = 0.f, s2 = 0.f;
  for (int j = tid; j < HW; j += 256) {
    const float e = __expf(Ar[j] - cmax[j]) * cscale[j];
    s1 += e * beta[j]; s2 += e * gama[j];
  }
  r1[tid] = s1; r2[tid] = s2; __syncthreads();
  for (int off = 128; off > 0; off >>= 1) {
    if (tid < off) { r1[tid] += r1[tid + off]; r2[tid] += r2[tid + off]; }
    __syncthreads();
  }
  if (tid == 0) { bh[i] = r1[0]; gh[i] = r2[0]; }
}

__global__ void amm_combine_kernel(const float* __restrict__ feat,
                                   const float* __restrict__ bh,
                                   const float* __restrict__ gh,
                                   float* __restrict__ out, int C, int HW) {
  const int idx = blockIdx.x * blockDim.x + threadIdx.x;
  if (idx >= C * HW) return;
  const int p = idx % HW;
  out[idx] = gh[p] * feat[idx] + bh[p];
}

__global__ void tanh_kernel(const float* __restrict__ in, float* __restrict__ out, int n) {
  const int idx = blockIdx.x * blockDim.x + threadIdx.x;
  if (idx < n) out[idx] = tanhf(in[idx]);
}

// ---------------------------------------------------------------------------
// Host side
// ---------------------------------------------------------------------------
static const char* kParamNames[89] = {
  "ab1","ab2","aw1","aw2","db0","db1","dc0","dc1","dc2","dg0","dg1",
  "dr0_b1","dr0_b2","dr0_c1","dr0_c2","dr0_g1","dr0_g2",
  "dr1_b1","dr1_b2","dr1_c1","dr1_c2","dr1_g1","dr1_g2",
  "dr2_b1","dr2_b2","dr2_c1","dr2_c2","dr2_g1","dr2_g2",
  "dt0","dt1","dtb0","dtb1","dtg0","dtg1",
  "ex_b0","ex_b1","ex_b2","ex_c0","ex_c1","ex_c2","ex_g0","ex_g1","ex_g2",
  "ex_r0_b1","ex_r0_b2","ex_r0_c1","ex_r0_c2","ex_r0_g1","ex_r0_g2",
  "ex_r1_b1","ex_r1_b2","ex_r1_c1","ex_r1_c2","ex_r1_g1","ex_r1_g2",
  "ex_r2_b1","ex_r2_b2","ex_r2_c1","ex_r2_c2","ex_r2_g1","ex_r2_g2",
  "ey_b0","ey_b1","ey_b2","ey_c0","ey_c1","ey_c2","ey_g0","ey_g1","ey_g2",
  "ey_r0_b1","ey_r0_b2","ey_r0_c1","ey_r0_c2","ey_r0_g1","ey_r0_g2",
  "ey_r1_b1","ey_r1_b2","ey_r1_c1","ey_r1_c2","ey_r1_g1","ey_r1_g2",
  "ey_r2_b1","ey_r2_b2","ey_r2_c1","ey_r2_c2","ey_r2_g1","ey_r2_g2",
};

static const float* PP(void* const* d_in, const char* n) {
  for (int i = 0; i < 89; ++i)
    if (!strcmp(kParamNames[i], n)) return (const float*)d_in[6 + i];
  return (const float*)d_in[6];
}

static void conv_layer(hipStream_t s, const float* in, const float* w, __bf16* wp,
                       float* out, int Cin, int Cout, int Hin, int Win,
                       int Hout, int Wout, int k, int st, int pad, int mode) {
  const int Ktot = Cin * k * k, Kch = (Ktot + 31) / 32;
  const int NT = ((Cout + 63) / 64) * 4;
  const int total = Kch * NT * 512;
  pack_weights_kernel<<<dim3((total + 255) / 256), dim3(256), 0, s>>>(
      w, wp, Cin, Cout, k, mode, NT, total);
  const int M = Hout * Wout;
  dim3 b(128), g(M / 64, (Cout + 63) / 64);
  if (k == 7)
    conv_wmma_kernel<7, 0><<<g, b, 0, s>>>(in, wp, out, Cin, Cout, Hin, Win, Hout, Wout, st, pad);
  else if (k == 3)
    conv_wmma_kernel<3, 0><<<g, b, 0, s>>>(in, wp, out, Cin, Cout, Hin, Win, Hout, Wout, st, pad);
  else if (mode == 0)
    conv_wmma_kernel<4, 0><<<g, b, 0, s>>>(in, wp, out, Cin, Cout, Hin, Win, Hout, Wout, st, pad);
  else
    conv_wmma_kernel<4, 1><<<g, b, 0, s>>>(in, wp, out, Cin, Cout, Hin, Win, Hout, Wout, st, pad);
}

static void inorm(hipStream_t s, const float* x, const float* g, const float* b,
                  const float* res, float* out, int C, int HW, int relu) {
  inorm_kernel<<<dim3(C), dim3(256), 0, s>>>(x, g, b, res, out, HW, relu);
}

// Workspace layout (float offsets), ~127 MB total
#define OFF_P0   ((size_t)0)
#define OFF_P1   ((size_t)4194304)
#define OFF_T0   ((size_t)8388608)
#define OFF_T1   ((size_t)9437184)
#define OFF_FX   ((size_t)10485760)
#define OFF_FY   ((size_t)11534336)
#define OFF_ATT  ((size_t)12582912)   // 16777216 floats
#define OFF_AB   ((size_t)29360128)   // 4096*416 bf16 = 851968 floats
#define OFF_BB   ((size_t)30212096)   // 13*256*512 bf16 = 851968 floats
#define OFF_WP   ((size_t)31064064)   // 1M bf16 = 524288 floats
#define OFF_BETA ((size_t)31588352)
#define OFF_GAMA ((size_t)31592448)
#define OFF_BH   ((size_t)31596544)
#define OFF_GH   ((size_t)31600640)
#define OFF_CMAX ((size_t)31604736)
#define OFF_CSC  ((size_t)31608832)

static void encoder_run(hipStream_t s, void* const* d_in, const float* x,
                        const char* pfx, float* P0, float* P1, float* T0,
                        float* T1, float* F, __bf16* WPk) {
  char nm[32];
  auto W = [&](const char* suf) {
    snprintf(nm, sizeof(nm), "%s%s", pfx, suf);
    return PP(d_in, nm);
  };
  conv_layer(s, x, W("c0"), WPk, P0, 3, 64, 256, 256, 256, 256, 7, 1, 3, 0);
  inorm(s, P0, W("g0"), W("b0"), nullptr, P0, 64, 65536, 1);
  conv_layer(s, P0, W("c1"), WPk, P1, 64, 128, 256, 256, 128, 128, 4, 2, 1, 0);
  inorm(s, P1, W("g1"), W("b1"), nullptr, P1, 128, 16384, 1);
  conv_layer(s, P1, W("c2"), WPk, F, 128, 256, 128, 128, 64, 64, 4, 2, 1, 0);
  inorm(s, F, W("g2"), W("b2"), nullptr, F, 256, 4096, 1);
  for (int r = 0; r < 3; ++r) {
    char suf[16];
    snprintf(suf, sizeof(suf), "r%d_c1", r); const float* c1 = W(suf);
    snprintf(suf, sizeof(suf), "r%d_g1", r); const float* g1 = W(suf);
    snprintf(suf, sizeof(suf), "r%d_b1", r); const float* b1 = W(suf);
    snprintf(suf, sizeof(suf), "r%d_c2", r); const float* c2 = W(suf);
    snprintf(suf, sizeof(suf), "r%d_g2", r); const float* g2 = W(suf);
    snprintf(suf, sizeof(suf), "r%d_b2", r); const float* b2 = W(suf);
    conv_layer(s, F, c1, WPk, T0, 256, 256, 64, 64, 64, 64, 3, 1, 1, 0);
    inorm(s, T0, g1, b1, nullptr, T0, 256, 4096, 1);
    conv_layer(s, T0, c2, WPk, T1, 256, 256, 64, 64, 64, 64, 3, 1, 1, 0);
    inorm(s, T1, g2, b2, F, F, 256, 4096, 0);   // F = F + inorm(conv2)
  }
}

extern "C" void kernel_launch(void* const* d_in, const int* in_sizes, int n_in,
                              void* d_out, int out_size, void* d_ws, size_t ws_size,
                              hipStream_t stream) {
  (void)in_sizes; (void)n_in; (void)out_size; (void)ws_size;
  const float* x    = (const float*)d_in[0];
  const float* y    = (const float*)d_in[1];
  const float* lm_x = (const float*)d_in[2];
  const float* lm_y = (const float*)d_in[3];
  const int*   mk_x = (const int*)d_in[4];
  const int*   mk_y = (const int*)d_in[5];

  float* ws = (float*)d_ws;
  float *P0 = ws + OFF_P0, *P1 = ws + OFF_P1;
  float *T0 = ws + OFF_T0, *T1 = ws + OFF_T1;
  float *FX = ws + OFF_FX, *FY = ws + OFF_FY, *AT = ws + OFF_ATT;
  __bf16 *AB  = (__bf16*)(ws + OFF_AB);
  __bf16 *BB  = (__bf16*)(ws + OFF_BB);
  __bf16 *WPk = (__bf16*)(ws + OFF_WP);
  float *BE = ws + OFF_BETA, *GA = ws + OFF_GAMA, *BH = ws + OFF_BH, *GH = ws + OFF_GH;
  float *CMX = ws + OFF_CMAX, *CSC = ws + OFF_CSC;

  // ---------------- encoders ----------------
  encoder_run(stream, d_in, x, "ex_", P0, P1, T0, T1, FX, WPk);
  encoder_run(stream, d_in, y, "ey_", P0, P1, T0, T1, FY, WPk);

  // ---------------- AMM attention ----------------
  const int HW = 4096, Kpad = 416, Kch = Kpad / 32, NTb = 256;
  betagamma_kernel<<<dim3((HW + 255) / 256), dim3(256), 0, stream>>>(
      FY, PP(d_in, "aw1"), PP(d_in, "ab1"), PP(d_in, "aw2"), PP(d_in, "ab2"),
      BE, GA, 256, HW);
  build_catA_kernel<<<dim3((HW * Kpad + 255) / 256), dim3(256), 0, stream>>>(
      FX, lm_x, AB, HW, Kpad);
  {
    const int total = Kch * NTb * 512;
    build_catB_frag_kernel<<<dim3((total + 255) / 256), dim3(256), 0, stream>>>(
        FY, lm_y, BB, HW, NTb, total);
  }
  {
    dim3 b(128), g(HW / 64, HW / 64);
    gemm_bf16_wmma_kernel<<<g, b, 0, stream>>>(AB, BB, AT, HW, HW, Kch, Kpad);
  }
  colstats_kernel<<<dim3(HW), dim3(256), 0, stream>>>(AT, mk_x, mk_y, CMX, CSC, HW);
  attvec_fused_kernel<<<dim3(HW), dim3(256), 0, stream>>>(AT, CMX, CSC, BE, GA, BH, GH, HW);
  amm_combine_kernel<<<dim3((256 * HW + 255) / 256), dim3(256), 0, stream>>>(
      FX, BH, GH, T0, 256, HW);   // T0 = feat_x_hat

  // ---------------- decoder ----------------
  for (int r = 0; r < 3; ++r) {
    char nm[32];
    auto D = [&](const char* fmt) {
      snprintf(nm, sizeof(nm), fmt, r);
      return PP(d_in, nm);
    };
    const float *c1 = D("dr%d_c1"), *g1 = D("dr%d_g1"), *b1 = D("dr%d_b1");
    const float *c2 = D("dr%d_c2"), *g2 = D("dr%d_g2"), *b2 = D("dr%d_b2");
    conv_layer(stream, T0, c1, WPk, T1, 256, 256, 64, 64, 64, 64, 3, 1, 1, 0);
    inorm(stream, T1, g1, b1, nullptr, T1, 256, 4096, 1);
    conv_layer(stream, T1, c2, WPk, FX, 256, 256, 64, 64, 64, 64, 3, 1, 1, 0);
    inorm(stream, FX, g2, b2, T0, T0, 256, 4096, 0);
  }
  // deconv dt0: (256,64,64) -> (128,128,128)
  conv_layer(stream, T0, PP(d_in, "dt0"), WPk, P0, 256, 128, 64, 64, 128, 128, 4, 1, 2, 1);
  inorm(stream, P0, PP(d_in, "dtg0"), PP(d_in, "dtb0"), nullptr, P0, 128, 16384, 1);
  // deconv dt1: (128,128,128) -> (64,256,256)
  conv_layer(stream, P0, PP(d_in, "dt1"), WPk, P1, 128, 64, 128, 128, 256, 256, 4, 1, 2, 1);
  inorm(stream, P1, PP(d_in, "dtg1"), PP(d_in, "dtb1"), nullptr, P1, 64, 65536, 1);
  // dc0, dc1 3x3 convs @64ch 256x256
  conv_layer(stream, P1, PP(d_in, "dc0"), WPk, P0, 64, 64, 256, 256, 256, 256, 3, 1, 1, 0);
  inorm(stream, P0, PP(d_in, "dg0"), PP(d_in, "db0"), nullptr, P0, 64, 65536, 1);
  conv_layer(stream, P0, PP(d_in, "dc1"), WPk, P1, 64, 64, 256, 256, 256, 256, 3, 1, 1, 0);
  inorm(stream, P1, PP(d_in, "dg1"), PP(d_in, "db1"), nullptr, P1, 64, 65536, 1);
  // dc2 7x7 -> 3 channels, then tanh to output
  conv_layer(stream, P1, PP(d_in, "dc2"), WPk, P0, 64, 3, 256, 256, 256, 256, 7, 1, 3, 0);
  tanh_kernel<<<dim3((3 * 65536 + 255) / 256), dim3(256), 0, stream>>>(
      P0, (float*)d_out, 3 * 65536);
}
